// HeteroUnsupervised_11742440587936
// MI455X (gfx1250) — matrix operand order repaired and checked by reading, not compiled, not run on hardware
//
#include <hip/hip_runtime.h>
#include <hip/hip_bf16.h>

typedef __attribute__((ext_vector_type(16))) _Float16 v16h;
typedef __attribute__((ext_vector_type(8)))  float    v8f;

#define N_NODES 100000
#define N_EDGES 800000
#define N_META  900000   // E + N self loops
#define D_IN    128
#define D_HID   128
#define D_OUTF  64

#define USE_ASYNC_LDS 1

// ---------------------------------------------------------------- utilities

__global__ void fill_u32(unsigned int* __restrict__ p, unsigned int v, int n) {
    int i = blockIdx.x * 256 + threadIdx.x;
    if (i < n) p[i] = v;
}

// Pack weight matrix W [C][128] (row-major f32) into WMMA-native B fragments:
// frag f = (nt*4 + kt)*32 + lane holds 16 contiguous halves, half h = W[(nt*16+lane%16)][kt*32 + (h&7) + 8*(lane/16) + 16*(h/8)].
// GEMM then loads each fragment as one aligned 32-byte vector (2 x global_load_b128).
__global__ void pack_b_frag(const float* __restrict__ W, _Float16* __restrict__ out, int C) {
    int o = blockIdx.x * 256 + threadIdx.x;    // C*128 outputs
    if (o >= C * 128) return;
    int h    = o & 15;
    int lane = (o >> 4) & 31;
    int kt   = (o >> 9) & 3;
    int nt   = o >> 11;
    out[o] = (_Float16)W[(nt * 16 + (lane & 15)) * 128 + kt * 32 + (h & 7) + 8 * (lane >> 4) + 16 * (h >> 3)];
}

// float atomic max via signed/unsigned int ordering trick (works across sign mix)
__device__ __forceinline__ void atomic_max_f32(float* addr, float v) {
    if (v >= 0.0f) atomicMax((int*)addr, __float_as_int(v));
    else           atomicMin((unsigned int*)addr, __float_as_uint(v));
}

// ---------------------------------------------------------------- SAGE aggregate
// agg[dst] += x_paper[src] ; cnt[dst] += 1   (32 lanes per edge, float4 each; L2-resident atomics)
__global__ __launch_bounds__(256)
void sage_aggregate(const float* __restrict__ xp,
                    const int* __restrict__ src, const int* __restrict__ dst,
                    float* __restrict__ agg, float* __restrict__ cnt) {
    int e    = blockIdx.x * 8 + (threadIdx.x >> 5);
    int lane = threadIdx.x & 31;
    if (e >= N_EDGES) return;
    int s = src[e], d = dst[e];
    float4 v = *(const float4*)(xp + (size_t)s * D_IN + lane * 4);
    float* base = agg + (size_t)d * D_IN + lane * 4;
    atomicAdd(base + 0, v.x); atomicAdd(base + 1, v.y);
    atomicAdd(base + 2, v.z); atomicAdd(base + 3, v.w);
    if (lane == 0) atomicAdd(cnt + d, 1.0f);
}

__global__ void p2a_scatter(const int* __restrict__ src, const int* __restrict__ dst,
                            int* __restrict__ p2a) {
    int e = blockIdx.x * 256 + threadIdx.x;
    if (e < N_EDGES) p2a[src[e]] = dst[e];   // duplicate-index winner unspecified (matches XLA scatter)
}

__global__ void meta_build(const int* __restrict__ ap_dst, const int* __restrict__ p2a,
                           int* __restrict__ meta_dst) {
    int e = blockIdx.x * 256 + threadIdx.x;
    if (e < N_EDGES) meta_dst[e] = p2a[ap_dst[e]];
}

// ---------------------------------------------------------------- WMMA fragment helpers
// 16-bit A-matrix 16x32 layout: lane L -> row M=L%16, group g=L/16;
// v16h half-pair j holds K = 2j + (j>=4 ? 8 : 0) + 8g (+0/+1)  -> coalesces into ds_load_b128
__device__ __forceinline__ v16h load_a_frag(const _Float16* __restrict__ lds_row_major,
                                            int M, int g, int k0) {
    v16h a;
#pragma unroll
    for (int j = 0; j < 8; ++j) {
        int K = k0 + 2 * j + ((j >= 4) ? 8 : 0) + 8 * g;
        a[2 * j]     = lds_row_major[M * 128 + K];
        a[2 * j + 1] = lds_row_major[M * 128 + K + 1];
    }
    return a;
}

// ---------------------------------------------------------------- SAGE linear (WMMA)
// h16[n][c] = leakyrelu_0.01( mean[n]@Wl.T + x_author[n]@Wr.T + b )   c in [0,128)
__global__ __launch_bounds__(256)
void sage_linear_wmma(const float* __restrict__ agg, const float* __restrict__ cnt,
                      const float* __restrict__ x_author,
                      const _Float16* __restrict__ WlPk, const _Float16* __restrict__ WrPk,
                      const float* __restrict__ bias,
                      _Float16* __restrict__ h16) {
    __shared__ _Float16 Am[16 * 128];   // mean tile
    __shared__ _Float16 Ax[16 * 128];   // root tile
    const int row0 = blockIdx.x * 16;
    for (int idx = threadIdx.x; idx < 16 * 128; idx += 256) {
        int r = idx >> 7, k = idx & 127;
        int n = row0 + r;
        float rc = 1.0f / fmaxf(cnt[n], 1.0f);
        Am[idx] = (_Float16)(agg[(size_t)n * 128 + k] * rc);
        Ax[idx] = (_Float16)(x_author[(size_t)n * 128 + k]);
    }
    __syncthreads();
    const int wave = threadIdx.x >> 5;     // = col tile nt, 8 waves x 16 = 128 cols
    const int lane = threadIdx.x & 31;
    const int M = lane & 15, g = lane >> 4;
    const v16h* Bl = (const v16h*)WlPk;
    const v16h* Br = (const v16h*)WrPk;
    v8f acc = {};
#pragma unroll
    for (int kt = 0; kt < 4; ++kt) {
        v16h a = load_a_frag(Am, M, g, kt * 32);
        v16h b = Bl[(wave * 4 + kt) * 32 + lane];
        acc = __builtin_amdgcn_wmma_f32_16x16x32_f16(false, a, false, b, (short)0, acc, false, false);
    }
#pragma unroll
    for (int kt = 0; kt < 4; ++kt) {
        v16h a = load_a_frag(Ax, M, g, kt * 32);
        v16h b = Br[(wave * 4 + kt) * 32 + lane];
        acc = __builtin_amdgcn_wmma_f32_16x16x32_f16(false, a, false, b, (short)0, acc, false, false);
    }
    const int col = wave * 16 + (lane & 15);
    const float bc = bias[col];
#pragma unroll
    for (int r = 0; r < 8; ++r) {
        int row = row0 + r + 8 * g;
        float v = acc[r] + bc;
        v = (v >= 0.0f) ? v : 0.01f * v;
        h16[(size_t)row * 128 + col] = (_Float16)v;
    }
}

// ---------------------------------------------------------------- GAT linear (WMMA)
// hout[n][c] = h16[rowsrc(n)] @ W.T ,  c in [0,64).  A tile staged via async global->LDS DMA.
__global__ __launch_bounds__(128)
void gat_gemm_wmma(const _Float16* __restrict__ h16, const _Float16* __restrict__ WPk,
                   const int* __restrict__ perm, int use_perm,
                   float* __restrict__ hout) {
    __shared__ _Float16 Am[16 * 128];
    const int row0 = blockIdx.x * 16;
#if USE_ASYNC_LDS
    // 16 rows x 256 B = 256 x 16-byte chunks; 128 threads -> 2 async b128 copies each.
    for (int chunk = threadIdx.x; chunk < 256; chunk += 128) {
        int r = chunk >> 4, seg = chunk & 15;
        int n = row0 + r;
        int sr = use_perm ? perm[n] : n;
        const _Float16* g = h16 + (size_t)sr * 128 + seg * 8;
        unsigned lds = (unsigned)(size_t)&Am[r * 128 + seg * 8];
        asm volatile("global_load_async_to_lds_b128 %0, %1, off"
                     :: "v"(lds), "v"(g) : "memory");
    }
    asm volatile("s_wait_asynccnt 0" ::: "memory");
    __syncthreads();
#else
    for (int idx = threadIdx.x; idx < 16 * 128; idx += 128) {
        int r = idx >> 7, k = idx & 127;
        int n = row0 + r;
        int sr = use_perm ? perm[n] : n;
        Am[idx] = h16[(size_t)sr * 128 + k];
    }
    __syncthreads();
#endif
    const int wave = threadIdx.x >> 5;     // col tile, 4 waves x 16 = 64 cols
    const int lane = threadIdx.x & 31;
    const int M = lane & 15, g = lane >> 4;
    const v16h* Bw = (const v16h*)WPk;
    v8f acc = {};
#pragma unroll
    for (int kt = 0; kt < 4; ++kt) {
        v16h a = load_a_frag(Am, M, g, kt * 32);
        v16h b = Bw[(wave * 4 + kt) * 32 + lane];
        acc = __builtin_amdgcn_wmma_f32_16x16x32_f16(false, a, false, b, (short)0, acc, false, false);
    }
    const int col = wave * 16 + (lane & 15);
#pragma unroll
    for (int r = 0; r < 8; ++r)
        hout[(size_t)(row0 + r + 8 * g) * 64 + col] = acc[r];
}

// ---------------------------------------------------------------- attention logits
__global__ __launch_bounds__(256)
void gat_alpha(const float* __restrict__ h, const float* __restrict__ a_src,
               const float* __restrict__ a_dst,
               float* __restrict__ asrc, float* __restrict__ adst) {
    int n    = blockIdx.x * 8 + (threadIdx.x >> 5);
    int lane = threadIdx.x & 31;
    if (n >= N_NODES) return;
    float h0 = h[(size_t)n * 64 + lane];
    float h1 = h[(size_t)n * 64 + 32 + lane];
    float s = h0 * a_src[lane] + h1 * a_src[32 + lane];
    float d = h0 * a_dst[lane] + h1 * a_dst[32 + lane];
#pragma unroll
    for (int o = 16; o; o >>= 1) { s += __shfl_xor(s, o); d += __shfl_xor(d, o); }
    if (lane == 0) { asrc[n] = s; adst[n] = d; }
}

__device__ __forceinline__ void meta_edge(int e, const int* ap_src, const int* meta_dst,
                                          int& s, int& d) {
    if (e < N_EDGES) { s = ap_src[e]; d = meta_dst[e]; }
    else             { s = e - N_EDGES; d = s; }
}

__global__ void gat_edge_max(const int* __restrict__ ap_src, const int* __restrict__ meta_dst,
                             const float* __restrict__ asrc, const float* __restrict__ adst,
                             float* __restrict__ segmax) {
    int e = blockIdx.x * 256 + threadIdx.x;
    if (e >= N_META) return;
    int s, d; meta_edge(e, ap_src, meta_dst, s, d);
    float v = asrc[s] + adst[d];
    v = (v >= 0.0f) ? v : 0.2f * v;
    atomic_max_f32(segmax + d, v);
}

__global__ void gat_edge_expsum(const int* __restrict__ ap_src, const int* __restrict__ meta_dst,
                                const float* __restrict__ asrc, const float* __restrict__ adst,
                                const float* __restrict__ segmax,
                                float* __restrict__ ex, float* __restrict__ den) {
    int e = blockIdx.x * 256 + threadIdx.x;
    if (e >= N_META) return;
    int s, d; meta_edge(e, ap_src, meta_dst, s, d);
    float v = asrc[s] + adst[d];
    v = (v >= 0.0f) ? v : 0.2f * v;
    float x = __expf(v - segmax[d]);
    ex[e] = x;
    atomicAdd(den + d, x);
}

__global__ __launch_bounds__(256)
void gat_edge_scatter(const int* __restrict__ ap_src, const int* __restrict__ meta_dst,
                      const float* __restrict__ ex, const float* __restrict__ den,
                      const float* __restrict__ h, float* __restrict__ out) {
    int e    = blockIdx.x * 16 + (threadIdx.x >> 4);
    int lane = threadIdx.x & 15;
    if (e >= N_META) return;
    int s, d; meta_edge(e, ap_src, meta_dst, s, d);
    float coef = ex[e] / den[d];
    float4 hv = *(const float4*)(h + (size_t)s * 64 + lane * 4);
    float* base = out + (size_t)d * 64 + lane * 4;
    atomicAdd(base + 0, coef * hv.x); atomicAdd(base + 1, coef * hv.y);
    atomicAdd(base + 2, coef * hv.z); atomicAdd(base + 3, coef * hv.w);
}

__global__ void gat_finalize(float* __restrict__ out2, const float* __restrict__ b,
                             const float* __restrict__ prelu_a) {
    int i = blockIdx.x * 256 + threadIdx.x;
    if (i >= 2 * N_NODES * 64) return;
    int c = i & 63;
    float v = out2[i] + b[c];
    float a = prelu_a[0];
    out2[i] = (v >= 0.0f) ? v : a * v;
}

__global__ __launch_bounds__(256)
void summary_mean(const float* __restrict__ pos, float* __restrict__ summary) {
    __shared__ float red[256];
    int c = blockIdx.x;
    float s = 0.0f;
    for (int n = threadIdx.x; n < N_NODES; n += 256) s += pos[(size_t)n * 64 + c];
    red[threadIdx.x] = s;
    __syncthreads();
    for (int o = 128; o; o >>= 1) {
        if (threadIdx.x < o) red[threadIdx.x] += red[threadIdx.x + o];
        __syncthreads();
    }
    if (threadIdx.x == 0) summary[c] = red[0] / (float)N_NODES;
}

// ---------------------------------------------------------------- launch

extern "C" void kernel_launch(void* const* d_in, const int* in_sizes, int n_in,
                              void* d_out, int out_size, void* d_ws, size_t ws_size,
                              hipStream_t stream) {
    const float* x_author = (const float*)d_in[0];
    const float* x_paper  = (const float*)d_in[1];
    const float* sage_Wl  = (const float*)d_in[2];   // [2][2][128][128]
    const float* sage_Wr  = (const float*)d_in[3];
    const float* sage_b   = (const float*)d_in[4];   // [2][2][128]
    const float* gat_W    = (const float*)d_in[5];   // [64][128]
    const float* gat_as   = (const float*)d_in[6];
    const float* gat_ad   = (const float*)d_in[7];
    const float* gat_b    = (const float*)d_in[8];
    const float* prelu_a  = (const float*)d_in[9];
    const int*   edge_ap  = (const int*)d_in[10];    // [2][E]
    const int*   edge_pa  = (const int*)d_in[11];
    const int*   perm     = (const int*)d_in[12];

    char* ws = (char*)d_ws;
    size_t off = 0;
    auto take = [&](size_t bytes) -> char* {
        char* p = ws + off;
        off += (bytes + 255) & ~(size_t)255;
        return p;
    };
    _Float16* WlPk   = (_Float16*)take(128 * 128 * 2);
    _Float16* WrPk   = (_Float16*)take(128 * 128 * 2);
    _Float16* WPk    = (_Float16*)take(64 * 128 * 2);
    float* agg       = (float*)take((size_t)N_NODES * 128 * 4);
    float* cnt       = (float*)take((size_t)N_NODES * 4);
    int*   p2a       = (int*)take((size_t)N_NODES * 4);
    int*   meta_dst  = (int*)take((size_t)N_EDGES * 4);
    _Float16* h16    = (_Float16*)take((size_t)N_NODES * 128 * 2);
    float* hpos      = (float*)take((size_t)N_NODES * 64 * 4);
    float* hneg      = (float*)take((size_t)N_NODES * 64 * 4);
    float* asrc_p    = (float*)take((size_t)N_NODES * 4);
    float* adst_p    = (float*)take((size_t)N_NODES * 4);
    float* asrc_n    = (float*)take((size_t)N_NODES * 4);
    float* adst_n    = (float*)take((size_t)N_NODES * 4);
    float* smax_p    = (float*)take((size_t)N_NODES * 4);
    float* smax_n    = (float*)take((size_t)N_NODES * 4);
    float* den_p     = (float*)take((size_t)N_NODES * 4);
    float* den_n     = (float*)take((size_t)N_NODES * 4);
    float* ex_p      = (float*)take((size_t)N_META * 4);
    float* ex_n      = (float*)take((size_t)N_META * 4);

    float* out_pos = (float*)d_out;
    float* out_neg = out_pos + (size_t)N_NODES * 64;
    float* out_sum = out_pos + 2 * (size_t)N_NODES * 64;

    const int* pa_src = edge_pa;            const int* pa_dst = edge_pa + N_EDGES;
    const int* ap_src = edge_ap;            const int* ap_dst = edge_ap + N_EDGES;

    auto G = [](int n) { return (n + 255) / 256; };

    // ---- init
    fill_u32<<<G(N_NODES * 128), 256, 0, stream>>>((unsigned*)agg, 0u, N_NODES * 128);
    fill_u32<<<G(N_NODES), 256, 0, stream>>>((unsigned*)cnt, 0u, N_NODES);
    fill_u32<<<G(N_NODES), 256, 0, stream>>>((unsigned*)p2a, 0u, N_NODES);
    fill_u32<<<G(N_NODES), 256, 0, stream>>>((unsigned*)smax_p, 0xFF800000u, N_NODES);
    fill_u32<<<G(N_NODES), 256, 0, stream>>>((unsigned*)smax_n, 0xFF800000u, N_NODES);
    fill_u32<<<G(N_NODES), 256, 0, stream>>>((unsigned*)den_p, 0u, N_NODES);
    fill_u32<<<G(N_NODES), 256, 0, stream>>>((unsigned*)den_n, 0u, N_NODES);
    fill_u32<<<G(2 * N_NODES * 64), 256, 0, stream>>>((unsigned*)out_pos, 0u, 2 * N_NODES * 64);

    // ---- weight pack (only layer l=1, edge_type=1 survives the reference's loop)
    pack_b_frag<<<G(128 * 128), 256, 0, stream>>>(sage_Wl + 3 * 128 * 128, WlPk, 128);
    pack_b_frag<<<G(128 * 128), 256, 0, stream>>>(sage_Wr + 3 * 128 * 128, WrPk, 128);
    pack_b_frag<<<G(64 * 128), 256, 0, stream>>>(gat_W, WPk, 64);

    // ---- SAGE aggregate + meta-path
    sage_aggregate<<<N_EDGES / 8, 256, 0, stream>>>(x_paper, pa_src, pa_dst, agg, cnt);
    p2a_scatter<<<G(N_EDGES), 256, 0, stream>>>(pa_src, pa_dst, p2a);
    meta_build<<<G(N_EDGES), 256, 0, stream>>>(ap_dst, p2a, meta_dst);

    // ---- SAGE linear (WMMA) -> h_author (f16)
    sage_linear_wmma<<<N_NODES / 16, 256, 0, stream>>>(agg, cnt, x_author, WlPk, WrPk,
                                                       sage_b + 3 * 128, h16);

    // ---- GAT dense lins (WMMA), pos + neg
    gat_gemm_wmma<<<N_NODES / 16, 128, 0, stream>>>(h16, WPk, perm, 0, hpos);
    gat_gemm_wmma<<<N_NODES / 16, 128, 0, stream>>>(h16, WPk, perm, 1, hneg);

    // ---- attention logits
    gat_alpha<<<N_NODES / 8, 256, 0, stream>>>(hpos, gat_as, gat_ad, asrc_p, adst_p);
    gat_alpha<<<N_NODES / 8, 256, 0, stream>>>(hneg, gat_as, gat_ad, asrc_n, adst_n);

    // ---- softmax over segments
    gat_edge_max<<<G(N_META), 256, 0, stream>>>(ap_src, meta_dst, asrc_p, adst_p, smax_p);
    gat_edge_max<<<G(N_META), 256, 0, stream>>>(ap_src, meta_dst, asrc_n, adst_n, smax_n);
    gat_edge_expsum<<<G(N_META), 256, 0, stream>>>(ap_src, meta_dst, asrc_p, adst_p, smax_p, ex_p, den_p);
    gat_edge_expsum<<<G(N_META), 256, 0, stream>>>(ap_src, meta_dst, asrc_n, adst_n, smax_n, ex_n, den_n);

    // ---- weighted scatter into outputs
    gat_edge_scatter<<<(N_META + 15) / 16, 256, 0, stream>>>(ap_src, meta_dst, ex_p, den_p, hpos, out_pos);
    gat_edge_scatter<<<(N_META + 15) / 16, 256, 0, stream>>>(ap_src, meta_dst, ex_n, den_n, hneg, out_neg);

    // ---- bias + PReLU, then summary
    gat_finalize<<<G(2 * N_NODES * 64), 256, 0, stream>>>(out_pos, gat_b, prelu_a);
    summary_mean<<<64, 256, 0, stream>>>(out_pos, out_sum);

    (void)in_sizes; (void)n_in; (void)out_size; (void)ws_size;
}
